// QwenVGGTInteractionv2_7894149890408
// MI455X (gfx1250) — matrix operand order repaired and checked by reading, not compiled, not run on hardware
//
#include <hip/hip_runtime.h>
#include <hip/hip_bf16.h>

typedef __bf16 bf16;
typedef __attribute__((ext_vector_type(16))) __bf16 v16bf;
typedef __attribute__((ext_vector_type(8)))  float  v8f;

#define D_MODEL 2048
#define NHEADS  16
#define HDIM    128
#define B_IMG   32
#define S_TOK   252
#define S_TOTAL (B_IMG * S_TOK)
#define HF      14
#define WF      18
#define IMP_H   512

union Frag {
    v16bf v;
    unsigned int u[8];
    bf16 h[16];
};

// low 32 bits of a flat LDS pointer == byte offset within the wave's LDS space
#define LDSOFF(p) ((unsigned)(uintptr_t)(p))

// CDNA5 async global->LDS copy (one 16B chunk per lane), tracked by ASYNCcnt.
static __device__ __forceinline__ void async_ld_b128(unsigned lds, const void* g) {
    asm volatile("global_load_async_to_lds_b128 %0, %1, off"
                 :: "v"(lds), "v"(g) : "memory");
}
static __device__ __forceinline__ void wait_async0() {
    asm volatile("s_wait_asynccnt 0x0" ::: "memory");
}

static __device__ __forceinline__ float wave_sum(float v) {
    #pragma unroll
    for (int o = 16; o > 0; o >>= 1) v += __shfl_down(v, o);
    return v;
}

// ---------------------------------------------------------------------------
// Kernel: fp32 -> bf16 conversion
// ---------------------------------------------------------------------------
__global__ __launch_bounds__(256) void cvt_kernel(const float* __restrict__ x,
                                                  bf16* __restrict__ y, int n) {
    int i = blockIdx.x * 256 + threadIdx.x;
    if (i < n) y[i] = (bf16)x[i];
}

// ---------------------------------------------------------------------------
// Kernel: LayerNorm (one block per token row) -> bf16
// ---------------------------------------------------------------------------
__global__ __launch_bounds__(256) void ln_kernel(const float* __restrict__ x,
                                                 const float* __restrict__ w,
                                                 const float* __restrict__ b,
                                                 bf16* __restrict__ y) {
    __shared__ float red[18];
    const int tid = threadIdx.x, lane = tid & 31, wv = tid >> 5;
    const float* row = x + (size_t)blockIdx.x * D_MODEL;
    float vals[8], s = 0.f, ss = 0.f;
    #pragma unroll
    for (int i = 0; i < 8; ++i) {
        vals[i] = row[tid + i * 256];
        s += vals[i];
        ss += vals[i] * vals[i];
    }
    s = wave_sum(s); ss = wave_sum(ss);
    if (lane == 0) { red[wv] = s; red[8 + wv] = ss; }
    __syncthreads();
    if (tid == 0) {
        float S = 0.f, SS = 0.f;
        #pragma unroll
        for (int j = 0; j < 8; ++j) { S += red[j]; SS += red[8 + j]; }
        float mu = S * (1.0f / D_MODEL);
        float var = SS * (1.0f / D_MODEL) - mu * mu;
        red[16] = mu;
        red[17] = rsqrtf(var + 1e-6f);
    }
    __syncthreads();
    float mu = red[16], rstd = red[17];
    bf16* yrow = y + (size_t)blockIdx.x * D_MODEL;
    #pragma unroll
    for (int i = 0; i < 8; ++i) {
        int c = tid + i * 256;
        yrow[c] = (bf16)((vals[i] - mu) * rstd * w[c] + b[c]);
    }
}

// ---------------------------------------------------------------------------
// Generic tiled WMMA GEMM:  C[M,N] = A[M,K](bf16) * W[N,K]^T(bf16)
// Block tile 128x64, 8 waves, each wave 32x32 (2x2 wmma 16x16 frags).
// Double-buffered LDS staged with global_load_async_to_lds_b128.
// Requires M%128==0, N%64==0, K%32==0 (true for all uses here).
// mode 0: C bf16    mode 1: C bf16, += bias[col], relu    mode 2: C f32 * tanh(aux[0])
// ---------------------------------------------------------------------------
__global__ __launch_bounds__(256) void gemm_bf16_kernel(
    const bf16* __restrict__ A, const bf16* __restrict__ W,
    void* __restrict__ Cout, int M, int N, int K,
    const float* __restrict__ aux, int mode)
{
    __shared__ __align__(16) bf16 As[2][128 * 32];
    __shared__ __align__(16) bf16 Bs[2][64 * 32];
    const int tid = threadIdx.x, lane = tid & 31, wv = tid >> 5;
    const int half = lane >> 4, l16 = lane & 15;
    const int m0 = blockIdx.y * 128, n0 = blockIdx.x * 64;
    const int wm = (wv >> 1) * 32, wn = (wv & 1) * 32;

    // per-thread staging coordinates
    const int ra = tid >> 1, ca = (tid & 1) * 16;    // A: 16 bf16 (2 x b128)
    const int rb = tid >> 2, cb = (tid & 3) * 8;     // W: 8 bf16 (1 x b128)
    const bf16* gA = A + (size_t)(m0 + ra) * K + ca;
    const bf16* gB = W + (size_t)(n0 + rb) * K + cb;

    auto stage = [&](int buf, int k0) {
        unsigned da = LDSOFF(&As[buf][ra * 32 + ca]);
        async_ld_b128(da,      gA + k0);
        async_ld_b128(da + 16, gA + k0 + 8);
        async_ld_b128(LDSOFF(&Bs[buf][rb * 32 + cb]), gB + k0);
    };

    v8f acc[2][2] = {};
    const int nk = K >> 5;
    stage(0, 0);
    for (int ki = 0; ki < nk; ++ki) {
        const int buf = ki & 1;
        wait_async0();
        __syncthreads();
        if (ki + 1 < nk) stage(buf ^ 1, (ki + 1) << 5);
        #pragma unroll
        for (int mt = 0; mt < 2; ++mt) {
            Frag a;
            const bf16* arow = &As[buf][(wm + mt * 16 + l16) * 32];
            #pragma unroll
            for (int i = 0; i < 8; ++i) {
                int p = ((i < 4) ? 2 * i : 2 * i + 8) + half * 8;
                a.u[i] = *(const unsigned int*)(arow + p);
            }
            #pragma unroll
            for (int nt = 0; nt < 2; ++nt) {
                Frag bf;
                const bf16* brow = &Bs[buf][(wn + nt * 16 + l16) * 32];
                #pragma unroll
                for (int i = 0; i < 8; ++i)
                    bf.u[i] = *(const unsigned int*)(brow + ((i < 4) ? 2 * i : 2 * i + 8) + half * 8);
                acc[mt][nt] = __builtin_amdgcn_wmma_f32_16x16x32_bf16(
                    false, a.v, false, bf.v, (short)0, acc[mt][nt], false, false);
            }
        }
    }

    float scale = 1.0f;
    if (mode == 2) scale = tanhf(aux[0]);
    #pragma unroll
    for (int mt = 0; mt < 2; ++mt)
    #pragma unroll
    for (int nt = 0; nt < 2; ++nt)
    #pragma unroll
    for (int v = 0; v < 8; ++v) {
        int row = m0 + wm + mt * 16 + half * 8 + v;
        int col = n0 + wn + nt * 16 + l16;
        float x = acc[mt][nt][v];
        if (mode == 1) x = fmaxf(x + aux[col], 0.0f);
        if (mode == 2)
            ((float*)Cout)[(size_t)row * N + col] = x * scale;
        else
            ((bf16*)Cout)[(size_t)row * N + col] = (bf16)x;
    }
}

// ---------------------------------------------------------------------------
// Kernel: per-token gate scores  gs[t,h] = sigmoid(sem_ln[t] . gate_w[h] + gb[h])
// ---------------------------------------------------------------------------
__global__ __launch_bounds__(256) void gate_kernel(const bf16* __restrict__ semln,
                                                   const bf16* __restrict__ gw,
                                                   const float* __restrict__ gb,
                                                   float* __restrict__ gs) {
    const int t = blockIdx.x, wv = threadIdx.x >> 5, lane = threadIdx.x & 31;
    const bf16* xr = semln + (size_t)t * D_MODEL;
    const bf16* wr = gw + (size_t)wv * D_MODEL;
    float s = 0.f;
    for (int k = lane; k < D_MODEL; k += 32) s += (float)xr[k] * (float)wr[k];
    s = wave_sum(s);
    if (lane == 0) gs[t * 8 + wv] = 1.0f / (1.0f + expf(-(s + gb[wv])));
}

// ---------------------------------------------------------------------------
// Kernel: log-importance  limp[t] = log(sigmoid(hidden[t].w2 + b2) + 0.1)
// ---------------------------------------------------------------------------
__global__ __launch_bounds__(256) void imp_kernel(const bf16* __restrict__ hidden,
                                                  const float* __restrict__ w2,
                                                  const float* __restrict__ b2,
                                                  float* __restrict__ limp) {
    const int t = blockIdx.x * 8 + (threadIdx.x >> 5);
    const int lane = threadIdx.x & 31;
    const bf16* hr = hidden + (size_t)t * IMP_H;
    float s = 0.f;
    for (int k = lane; k < IMP_H; k += 32) s += (float)hr[k] * w2[k];
    s = wave_sum(s);
    if (lane == 0) {
        float im = 1.0f / (1.0f + expf(-(s + b2[0])));
        limp[t] = logf(im + 0.1f);
    }
}

// ---------------------------------------------------------------------------
// Attention kernel. grid = (4 qtiles, 16 heads, 32 images), 256 threads.
// Scores in registers; K/V staged via async-to-LDS; probs staged bf16 in LDS.
// ---------------------------------------------------------------------------
__global__ __launch_bounds__(256) void attn_kernel(
    const bf16* __restrict__ Qb, const bf16* __restrict__ Kb,
    const bf16* __restrict__ Vb, const float* __restrict__ gs,
    const float* __restrict__ limp, bf16* __restrict__ Ob)
{
    __shared__ __align__(16) bf16 KVs[64 * 128];     // 16 KB (K then V staging)
    __shared__ __align__(16) bf16 Pb[64 * 256];      // 32 KB probs
    __shared__ float RedMax[2][64];
    __shared__ float RedSum[2][64];

    const int qt = blockIdx.x, h = blockIdx.y, b = blockIdx.z;
    const int tid = threadIdx.x, lane = tid & 31, wv = tid >> 5;
    const int half = lane >> 4, l16 = lane & 15;
    const int q0 = qt * 64;
    const int mw = wv & 3, kh = wv >> 2;   // wave: query 16-row tile, key half

    // staging coordinates (64 x 128 tile, 32 bf16 = 4 x b128 per thread)
    const int rs = tid >> 2, cs = (tid & 3) * 32;
    const unsigned kvdst = LDSOFF(&KVs[rs * 128 + cs]);

    auto stage_tile = [&](const bf16* __restrict__ src_base, int kt) {
        const int key = kt * 64 + rs;
        if (key < S_TOK) {
            const bf16* src = src_base + (size_t)(b * S_TOK + key) * D_MODEL + h * HDIM + cs;
            #pragma unroll
            for (int j = 0; j < 4; ++j)
                async_ld_b128(kvdst + j * 16, src + j * 8);
        } else {
            uint4 z = {0u, 0u, 0u, 0u};
            #pragma unroll
            for (int j = 0; j < 4; ++j)
                ((uint4*)(KVs + rs * 128 + cs))[j] = z;
        }
        wait_async0();
    };

    // ---- Q fragments held in registers (4 chunks of K=32 over HDIM=128) ----
    Frag aq[4];
    {
        const int tq = q0 + mw * 16 + l16;
        const bool valid = tq < S_TOK;
        const bf16* qrow = Qb + (size_t)(b * S_TOK + (valid ? tq : 0)) * D_MODEL + h * HDIM;
        #pragma unroll
        for (int kc = 0; kc < 4; ++kc)
        #pragma unroll
        for (int i = 0; i < 8; ++i) {
            int p = ((i < 4) ? 2 * i : 2 * i + 8) + half * 8 + kc * 32;
            aq[kc].u[i] = valid ? *(const unsigned int*)(qrow + p) : 0u;
        }
    }

    // ---- Phase 1: scores = Q K^T (wave covers 16 q-rows x 128 keys) ----
    v8f sacc[8] = {};
    for (int kt = 0; kt < 4; ++kt) {
        stage_tile(Kb, kt);
        __syncthreads();
        if ((kt >> 1) == kh) {
            #pragma unroll
            for (int kc = 0; kc < 4; ++kc) {
                #pragma unroll
                for (int nt = 0; nt < 4; ++nt) {
                    Frag bk;
                    const bf16* krow = KVs + (nt * 16 + l16) * 128 + kc * 32;
                    #pragma unroll
                    for (int i = 0; i < 8; ++i)
                        bk.u[i] = *(const unsigned int*)(krow + ((i < 4) ? 2 * i : 2 * i + 8) + half * 8);
                    const int f = (kt & 1) * 4 + nt;
                    sacc[f] = __builtin_amdgcn_wmma_f32_16x16x32_bf16(
                        false, aq[kc].v, false, bk.v, (short)0, sacc[f], false, false);
                }
            }
        }
        __syncthreads();
    }

    // ---- Epilogue: scale + spatial bias * gate + log importance + masking ----
    const float inv_sqrt = 0.0883883476483184f;                   // 1/sqrt(128)
    const float inv_diag = 1.0f / (sqrtf((float)(HF * HF + WF * WF)) + 1e-6f);
    float gsv[8];
    int   tqv[8];
    #pragma unroll
    for (int v = 0; v < 8; ++v) {
        int tq = q0 + mw * 16 + half * 8 + v;
        tqv[v] = tq;
        gsv[v] = (h < 8 && tq < S_TOK) ? gs[(b * S_TOK + tq) * 8 + h] : 0.0f;
    }
    #pragma unroll
    for (int f = 0; f < 8; ++f) {
        const int tk = (2 * kh + (f >> 2)) * 64 + (f & 3) * 16 + l16;
        const bool kval = tk < S_TOK;
        const float li = kval ? limp[b * S_TOK + tk] : 0.0f;
        const int ky = tk / WF, kx = tk - ky * WF;
        #pragma unroll
        for (int v = 0; v < 8; ++v) {
            float x = sacc[f][v] * inv_sqrt + li;
            const int tq = tqv[v];
            const int qy = tq / WF, qx = tq - qy * WF;
            const float dy = (float)(qy - ky), dx = (float)(qx - kx);
            x += gsv[v] * (-sqrtf(dy * dy + dx * dx) * inv_diag);
            if (!kval) x = -__builtin_huge_valf();
            sacc[f][v] = x;
        }
    }

    // ---- Phase 2: softmax over 252 keys (16-lane shfl + cross-wave LDS) ----
    float m8[8];
    #pragma unroll
    for (int v = 0; v < 8; ++v) {
        float mx = -__builtin_huge_valf();
        #pragma unroll
        for (int f = 0; f < 8; ++f) mx = fmaxf(mx, sacc[f][v]);
        #pragma unroll
        for (int o = 1; o < 16; o <<= 1) mx = fmaxf(mx, __shfl_xor(mx, o));
        m8[v] = mx;
    }
    if (l16 == 0) {
        #pragma unroll
        for (int v = 0; v < 8; ++v) RedMax[kh][mw * 16 + half * 8 + v] = m8[v];
    }
    __syncthreads();
    #pragma unroll
    for (int v = 0; v < 8; ++v) {
        int rl = mw * 16 + half * 8 + v;
        m8[v] = fmaxf(RedMax[0][rl], RedMax[1][rl]);
    }
    float s8[8];
    #pragma unroll
    for (int v = 0; v < 8; ++v) {
        float s = 0.f;
        #pragma unroll
        for (int f = 0; f < 8; ++f) {
            float ev = expf(sacc[f][v] - m8[v]);
            sacc[f][v] = ev;
            s += ev;
        }
        #pragma unroll
        for (int o = 1; o < 16; o <<= 1) s += __shfl_xor(s, o);
        s8[v] = s;
    }
    if (l16 == 0) {
        #pragma unroll
        for (int v = 0; v < 8; ++v) RedSum[kh][mw * 16 + half * 8 + v] = s8[v];
    }
    __syncthreads();
    // write normalized probs to LDS (bf16, row-major 64x256)
    #pragma unroll
    for (int v = 0; v < 8; ++v) {
        int rl = mw * 16 + half * 8 + v;
        float inv = 1.0f / (RedSum[0][rl] + RedSum[1][rl]);
        #pragma unroll
        for (int f = 0; f < 8; ++f) {
            int col = (2 * kh + (f >> 2)) * 64 + (f & 3) * 16 + l16;
            Pb[rl * 256 + col] = (bf16)(sacc[f][v] * inv);
        }
    }
    __syncthreads();

    // ---- Phase 3: out = P @ V  (wave: 16 q-rows x 64 head-dims) ----
    v8f oacc[4] = {};
    const int mw2 = wv & 3, nqh = (wv >> 2) * 64;
    for (int kt = 0; kt < 4; ++kt) {
        __syncthreads();
        stage_tile(Vb, kt);
        __syncthreads();
        #pragma unroll
        for (int kc = 0; kc < 2; ++kc) {
            Frag ap;
            const bf16* prow = Pb + (mw2 * 16 + l16) * 256 + kt * 64 + kc * 32;
            #pragma unroll
            for (int i = 0; i < 8; ++i) {
                int p = ((i < 4) ? 2 * i : 2 * i + 8) + half * 8;
                ap.u[i] = *(const unsigned int*)(prow + p);
            }
            #pragma unroll
            for (int nt = 0; nt < 4; ++nt) {
                Frag bv;
                const int nb = nqh + nt * 16 + l16;
                #pragma unroll
                for (int i = 0; i < 8; ++i) {
                    int k0 = ((i < 4) ? 2 * i : 2 * i + 8) + half * 8 + kc * 32;
                    bv.h[2 * i]     = KVs[k0 * 128 + nb];
                    bv.h[2 * i + 1] = KVs[(k0 + 1) * 128 + nb];
                }
                oacc[nt] = __builtin_amdgcn_wmma_f32_16x16x32_bf16(
                    false, ap.v, false, bv.v, (short)0, oacc[nt], false, false);
            }
        }
    }
    // ---- write attention output (bf16) ----
    #pragma unroll
    for (int nt = 0; nt < 4; ++nt)
    #pragma unroll
    for (int v = 0; v < 8; ++v) {
        const int tq = q0 + mw2 * 16 + half * 8 + v;
        if (tq < S_TOK)
            Ob[(size_t)(b * S_TOK + tq) * D_MODEL + h * HDIM + nqh + nt * 16 + l16] =
                (bf16)oacc[nt][v];
    }
}

// ---------------------------------------------------------------------------
// Host launcher
// ---------------------------------------------------------------------------
extern "C" void kernel_launch(void* const* d_in, const int* in_sizes, int n_in,
                              void* d_out, int out_size, void* d_ws, size_t ws_size,
                              hipStream_t stream) {
    const float* sem  = (const float*)d_in[0];
    const float* vgg  = (const float*)d_in[1];
    // d_in[2] = grid_thw (int64) -> static 28x36, pooled 14x18: hardcoded
    const float* ln1w = (const float*)d_in[3];
    const float* ln1b = (const float*)d_in[4];
    const float* ln2w = (const float*)d_in[5];
    const float* ln2b = (const float*)d_in[6];
    const float* Wq   = (const float*)d_in[7];
    const float* Wk   = (const float*)d_in[8];
    const float* Wv   = (const float*)d_in[9];
    const float* Wo   = (const float*)d_in[10];
    const float* iw1  = (const float*)d_in[11];
    const float* ib1  = (const float*)d_in[12];
    const float* iw2  = (const float*)d_in[13];
    const float* ib2  = (const float*)d_in[14];
    const float* gw   = (const float*)d_in[15];
    const float* gb   = (const float*)d_in[16];
    const float* gate = (const float*)d_in[17];

    char* ws = (char*)d_ws;
    size_t off = 0;
    auto take = [&](size_t bytes) { char* p = ws + off; off += (bytes + 255) & ~(size_t)255; return p; };

    const size_t SD  = (size_t)S_TOTAL * D_MODEL;  // 16.5M elems
    bf16* sem_ln = (bf16*)take(SD * 2);
    bf16* vgg_ln = (bf16*)take(SD * 2);            // reused as attn_out later
    bf16* q_buf  = (bf16*)take(SD * 2);
    bf16* k_buf  = (bf16*)take(SD * 2);
    bf16* v_buf  = (bf16*)take(SD * 2);
    bf16* wq_b   = (bf16*)take((size_t)D_MODEL * D_MODEL * 2);
    bf16* wk_b   = (bf16*)take((size_t)D_MODEL * D_MODEL * 2);
    bf16* wv_b   = (bf16*)take((size_t)D_MODEL * D_MODEL * 2);
    bf16* wo_b   = (bf16*)take((size_t)D_MODEL * D_MODEL * 2);
    bf16* iw1_b  = (bf16*)take((size_t)IMP_H * D_MODEL * 2);
    bf16* gw_b   = (bf16*)take((size_t)8 * D_MODEL * 2);
    bf16* hidden = (bf16*)take((size_t)S_TOTAL * IMP_H * 2);
    float* gscore = (float*)take((size_t)S_TOTAL * 8 * 4);
    float* limp   = (float*)take((size_t)S_TOTAL * 4);

    const int DD = D_MODEL * D_MODEL;
    // 1) weights -> bf16
    cvt_kernel<<<(DD + 255) / 256, 256, 0, stream>>>(Wq, wq_b, DD);
    cvt_kernel<<<(DD + 255) / 256, 256, 0, stream>>>(Wk, wk_b, DD);
    cvt_kernel<<<(DD + 255) / 256, 256, 0, stream>>>(Wv, wv_b, DD);
    cvt_kernel<<<(DD + 255) / 256, 256, 0, stream>>>(Wo, wo_b, DD);
    cvt_kernel<<<(IMP_H * D_MODEL + 255) / 256, 256, 0, stream>>>(iw1, iw1_b, IMP_H * D_MODEL);
    cvt_kernel<<<(8 * D_MODEL + 255) / 256, 256, 0, stream>>>(gw, gw_b, 8 * D_MODEL);

    // 2) layernorm -> bf16
    ln_kernel<<<S_TOTAL, 256, 0, stream>>>(sem, ln1w, ln1b, sem_ln);
    ln_kernel<<<S_TOTAL, 256, 0, stream>>>(vgg, ln2w, ln2b, vgg_ln);

    // 3) Q/K/V projections (M=8064, N=2048, K=2048)
    dim3 gqkv(D_MODEL / 64, S_TOTAL / 128);
    gemm_bf16_kernel<<<gqkv, 256, 0, stream>>>(sem_ln, wq_b, q_buf, S_TOTAL, D_MODEL, D_MODEL, nullptr, 0);
    gemm_bf16_kernel<<<gqkv, 256, 0, stream>>>(vgg_ln, wk_b, k_buf, S_TOTAL, D_MODEL, D_MODEL, nullptr, 0);
    gemm_bf16_kernel<<<gqkv, 256, 0, stream>>>(vgg_ln, wv_b, v_buf, S_TOTAL, D_MODEL, D_MODEL, nullptr, 0);

    // 4) gates + importance
    gate_kernel<<<S_TOTAL, 256, 0, stream>>>(sem_ln, gw_b, gb, gscore);
    dim3 ghid(IMP_H / 64, S_TOTAL / 128);
    gemm_bf16_kernel<<<ghid, 256, 0, stream>>>(sem_ln, iw1_b, hidden, S_TOTAL, IMP_H, D_MODEL, ib1, 1);
    imp_kernel<<<S_TOTAL / 8, 256, 0, stream>>>(hidden, iw2, ib2, limp);

    // 5) attention; output reuses vgg_ln buffer (vgg_ln fully consumed by K/V GEMMs)
    bf16* attn_out = vgg_ln;
    dim3 gattn(4, NHEADS, B_IMG);
    attn_kernel<<<gattn, 256, 0, stream>>>(q_buf, k_buf, v_buf, gscore, limp, attn_out);

    // 6) output projection, scaled by tanh(gate), f32 out
    gemm_bf16_kernel<<<gqkv, 256, 0, stream>>>(attn_out, wo_b, d_out, S_TOTAL, D_MODEL, D_MODEL, gate, 2);
}
